// MultiHeadRelativeAttention_78116865180296
// MI455X (gfx1250) — matrix operand order repaired and checked by reading, not compiled, not run on hardware
//
#include <hip/hip_runtime.h>

// ---- fixed problem shape (from reference) ----
#define LSEQ 2048
#define DM   512
#define NH   8
#define PDH  64
#define NB   2
#define RTOT 4096   // NB*LSEQ

typedef __attribute__((ext_vector_type(16))) __bf16 v16bf;
typedef __attribute__((ext_vector_type(8)))  float  v8f;
typedef unsigned short u16;

// fp32 -> bf16, round-to-nearest-even
static __device__ inline u16 f2bf(float f) {
    unsigned u = __builtin_bit_cast(unsigned, f);
    unsigned r = u + 0x7FFFu + ((u >> 16) & 1u);
    return (u16)(r >> 16);
}

struct Frag { union { v16bf v; uint4 q[2]; }; };

static __device__ inline v8f wmma_bf16(const Frag& a, const Frag& b, v8f c) {
    return __builtin_amdgcn_wmma_f32_16x16x32_bf16(
        /*neg_a*/false, a.v, /*neg_b*/false, b.v,
        /*c_mod*/(short)0, c, /*reuse_a*/false, /*reuse_b*/false);
}

// A-fragment (16x32 bf16, M x K): row = lane&15; lane<16 holds K 0..15,
// lane>=16 holds K 16..31  -> two 16B contiguous loads.
static __device__ inline Frag loadA(const u16* base, int ldk, int row0, int kOff, int lane) {
    int r  = row0 + (lane & 15);
    int kh = (lane >> 4) * 16;
    const u16* p = base + (size_t)r * ldk + kOff + kh;
    Frag f;
    f.q[0] = *reinterpret_cast<const uint4*>(p);        // K = kh..kh+7
    f.q[1] = *reinterpret_cast<const uint4*>(p + 8);    // K = kh+8..kh+15
    return f;
}

// B-fragment (32x16 bf16, K x N): col = lane&15; lane<16 holds K 0..15,
// lane>=16 K 16..31 -> one 32B contiguous load per lane (rows of length ldk,
// row index == output column, i.e. "transposed" storage).
static __device__ inline Frag loadB(const u16* base, int ldk, int col0, int kOff, int lane) {
    int c  = col0 + (lane & 15);
    int kg = (lane >> 4) * 16;
    const u16* p = base + (size_t)c * ldk + kOff + kg;
    Frag f;
    f.q[0] = *reinterpret_cast<const uint4*>(p);
    f.q[1] = *reinterpret_cast<const uint4*>(p + 8);
    return f;
}

// ---------------- conversion kernels ----------------
__global__ void cvt_bf16_kernel(const float* __restrict__ s, u16* __restrict__ d, int n) {
    int i = blockIdx.x * blockDim.x + threadIdx.x;
    if (i < n) d[i] = f2bf(s[i]);
}

// W [512x512] row-major (k-major) -> WT [n][k] bf16
__global__ void transpose_cvt_kernel(const float* __restrict__ W, u16* __restrict__ WT) {
    int t = blockIdx.x * blockDim.x + threadIdx.x;   // 0..262143
    int k = t >> 9, n = t & 511;
    WT[(size_t)n * DM + k] = f2bf(W[t]);
}

// ---------------- QKV projection ----------------
// Y = Xbf @ W + b ; z=0->q[B,H,L,64], z=1->k[B,H,L,64], z=2->vT[B,H,64,L]
__global__ __launch_bounds__(256) void qkv_proj_kernel(
    const u16* __restrict__ xbf, const u16* __restrict__ WT3,
    const float* __restrict__ bq, const float* __restrict__ bk, const float* __restrict__ bv,
    u16* __restrict__ qout, u16* __restrict__ kout, u16* __restrict__ vTout)
{
    const int lane = threadIdx.x & 31, wid = threadIdx.x >> 5;
    const int col = lane & 15, hi = lane >> 4;
    const int z  = blockIdx.z;
    const u16* WT = WT3 + (size_t)z * DM * DM;
    const float* bias = (z == 0) ? bq : (z == 1) ? bk : bv;
    const int r0 = (blockIdx.y * 8 + wid) * 16;
    const int c0 = blockIdx.x * 16;

    v8f acc = {};
    for (int kc = 0; kc < DM; kc += 32) {
        Frag a = loadA(xbf, DM, r0, kc, lane);
        Frag b = loadB(WT,  DM, c0, kc, lane);
        acc = wmma_bf16(a, b, acc);
    }
    const float bval = bias[c0 + col];
#pragma unroll
    for (int v = 0; v < 8; ++v) {
        int r = r0 + v + 8 * hi;
        int bb = r >> 11, l = r & (LSEQ - 1);
        int c = c0 + col;
        int h = c >> 6, pd = c & 63;
        u16 val = f2bf(acc[v] + bval);
        if (z == 0)      qout [(((size_t)bb * NH + h) * LSEQ + l) * PDH + pd] = val;
        else if (z == 1) kout [(((size_t)bb * NH + h) * LSEQ + l) * PDH + pd] = val;
        else             vTout[(((size_t)bb * NH + h) * PDH + pd) * LSEQ + l] = val;
    }
}

// ---------------- fused flash attention with relative bias ----------------
// rel[l,m] = q_l . E[L-1-(l-m)]  (non-causal elems are masked to -10000 anyway)
__global__ __launch_bounds__(256) void attn_kernel(
    const u16* __restrict__ qb, const u16* __restrict__ kb,
    const u16* __restrict__ vT, const u16* __restrict__ Eb,
    u16* __restrict__ attnbf)
{
    const int lane = threadIdx.x & 31, wid = threadIdx.x >> 5;
    const int col = lane & 15, hi = lane >> 4;
    const int bh = blockIdx.y;                 // b*NH + h
    const int l0 = (blockIdx.x * 8 + wid) * 16;

    const size_t head = (size_t)bh * LSEQ * PDH;
    const u16* Q = qb + head;
    const u16* K = kb + head;
    const u16* V = vT + head;                  // [64][L]

    __shared__ __align__(16) u16 pbuf[8][16 * 40];
    u16* pw = pbuf[wid];

    Frag qa0 = loadA(Q, PDH, l0, 0,  lane);    // d 0..31
    Frag qa1 = loadA(Q, PDH, l0, 32, lane);    // d 32..63

    v8f O[4] = {v8f{}, v8f{}, v8f{}, v8f{}};
    float mrow[8], lrow[8];
#pragma unroll
    for (int v = 0; v < 8; ++v) { mrow[v] = -1e30f; lrow[v] = 0.f; }

    const int nIter = (l0 >> 5) + 1;
    for (int it = 0; it < nIter; ++it) {
        const int m0 = it * 32;

        // ---- S = Q K^T (16 x 32) ----
        v8f S0 = {}, S1 = {};
        { Frag b;
          b = loadB(K, PDH, m0,      0,  lane); S0 = wmma_bf16(qa0, b, S0);
          b = loadB(K, PDH, m0,      32, lane); S0 = wmma_bf16(qa1, b, S0);
          b = loadB(K, PDH, m0 + 16, 0,  lane); S1 = wmma_bf16(qa0, b, S1);
          b = loadB(K, PDH, m0 + 16, 32, lane); S1 = wmma_bf16(qa1, b, S1); }

        // ---- T tiles over diagonal band of E: rows j0 .. j0+47 ----
        const int j0 = (LSEQ - 16) - l0 + m0;
        v8f T[3] = {v8f{}, v8f{}, v8f{}};
#pragma unroll
        for (int t = 0; t < 3; ++t) {
            int er = j0 + t * 16 + col;
            er = (er < LSEQ) ? er : (LSEQ - 1);        // clamp: only masked elems use OOB rows
            const u16* ep = Eb + (size_t)er * PDH + (hi ? 16 : 0);
            Frag b0; b0.q[0] = *reinterpret_cast<const uint4*>(ep);
                     b0.q[1] = *reinterpret_cast<const uint4*>(ep + 8);
            T[t] = wmma_bf16(qa0, b0, T[t]);
            const u16* ep1 = ep + 32;
            Frag b1; b1.q[0] = *reinterpret_cast<const uint4*>(ep1);
                     b1.q[1] = *reinterpret_cast<const uint4*>(ep1 + 8);
            T[t] = wmma_bf16(qa1, b1, T[t]);
        }

        // ---- scores: rel shuffle + scale + causal mask, then online softmax ----
        float s0[8], s1[8];
#pragma unroll
        for (int v = 0; v < 8; ++v) {
            const int dl  = v + 8 * hi;                 // l - l0
            const int idx = 15 + col - dl;              // 0..30, col-tile 0
            const int sl  = (idx & 15) + (hi << 4);
            float c0v = __shfl(T[0][v], sl, 32);
            float c1v = __shfl(T[1][v], sl, 32);
            float c2v = __shfl(T[2][v], sl, 32);
            float rel0 = (idx < 16) ? c0v : c1v;        // tile idx>>4
            float rel1 = (idx < 16) ? c1v : c2v;        // idx+16 -> tile 1 or 2
            const int l  = l0 + dl;
            const int ma = m0 + col, mb = ma + 16;
            s0[v] = (ma <= l) ? (S0[v] + rel0) * 0.125f : -10000.0f;
            s1[v] = (mb <= l) ? (S1[v] + rel1) * 0.125f : -10000.0f;
        }
#pragma unroll
        for (int v = 0; v < 8; ++v) {
            float mx = fmaxf(s0[v], s1[v]);
#pragma unroll
            for (int o = 1; o < 16; o <<= 1) mx = fmaxf(mx, __shfl_xor(mx, o, 32));
            float mnew  = fmaxf(mrow[v], mx);
            float scale = __expf(mrow[v] - mnew);
            mrow[v] = mnew;
            float p0 = __expf(s0[v] - mnew);
            float p1 = __expf(s1[v] - mnew);
            float rs = p0 + p1;
#pragma unroll
            for (int o = 1; o < 16; o <<= 1) rs += __shfl_xor(rs, o, 32);
            lrow[v] = lrow[v] * scale + rs;
            O[0][v] *= scale; O[1][v] *= scale; O[2][v] *= scale; O[3][v] *= scale;
            const int prow = v + 8 * hi;
            pw[prow * 40 + col]      = f2bf(p0);
            pw[prow * 40 + col + 16] = f2bf(p1);
        }

        // ---- A-fragment of P from LDS (wave-synchronous; DS ops in order) ----
        Frag pa;
        { const u16* pp = pw + (size_t)col * 40 + (hi ? 16 : 0);
          pa.q[0] = *reinterpret_cast<const uint4*>(pp);
          pa.q[1] = *reinterpret_cast<const uint4*>(pp + 8); }

        // ---- O += P V  (K = 32 keys) ----
#pragma unroll
        for (int dt = 0; dt < 4; ++dt) {
            const u16* vp = V + (size_t)(dt * 16 + col) * LSEQ + m0 + (hi ? 16 : 0);
            Frag b; b.q[0] = *reinterpret_cast<const uint4*>(vp);
                    b.q[1] = *reinterpret_cast<const uint4*>(vp + 8);
            O[dt] = wmma_bf16(pa, b, O[dt]);
        }
    }

    // ---- epilogue: normalize, store attn as bf16 [B, L, D] with D = h*64+d ----
    const int bb = bh >> 3, h = bh & 7;
#pragma unroll
    for (int v = 0; v < 8; ++v) {
        float inv = 1.0f / lrow[v];
        int l = l0 + v + 8 * hi;
        size_t base = ((size_t)(bb * LSEQ + l)) * DM + h * PDH;
#pragma unroll
        for (int dt = 0; dt < 4; ++dt)
            attnbf[base + dt * 16 + col] = f2bf(O[dt][v] * inv);
    }
}

// ---------------- output projection ----------------
__global__ __launch_bounds__(256) void out_proj_kernel(
    const u16* __restrict__ attnbf, const u16* __restrict__ WoT,
    const float* __restrict__ bo, float* __restrict__ out)
{
    const int lane = threadIdx.x & 31, wid = threadIdx.x >> 5;
    const int col = lane & 15, hi = lane >> 4;
    const int r0 = (blockIdx.y * 8 + wid) * 16;
    const int c0 = blockIdx.x * 16;

    v8f acc = {};
    for (int kc = 0; kc < DM; kc += 32) {
        Frag a = loadA(attnbf, DM, r0, kc, lane);
        Frag b = loadB(WoT,    DM, c0, kc, lane);
        acc = wmma_bf16(a, b, acc);
    }
    const float bval = bo[c0 + col];
#pragma unroll
    for (int v = 0; v < 8; ++v) {
        int r = r0 + v + 8 * hi;
        out[(size_t)r * DM + c0 + col] = acc[v] + bval;
    }
}

// ---------------- host launch ----------------
extern "C" void kernel_launch(void* const* d_in, const int* in_sizes, int n_in,
                              void* d_out, int out_size, void* d_ws, size_t ws_size,
                              hipStream_t stream) {
    (void)in_sizes; (void)n_in; (void)out_size; (void)ws_size;
    const float* x  = (const float*)d_in[0];
    const float* Wq = (const float*)d_in[1];
    const float* bq = (const float*)d_in[2];
    const float* Wk = (const float*)d_in[3];
    const float* bk = (const float*)d_in[4];
    const float* Wv = (const float*)d_in[5];
    const float* bv = (const float*)d_in[6];
    const float* Wo = (const float*)d_in[7];
    const float* bo = (const float*)d_in[8];
    const float* E  = (const float*)d_in[9];
    float* out = (float*)d_out;

    // workspace carve (all bf16 regions, 256B aligned)
    char* ws = (char*)d_ws;
    size_t off = 0;
    auto carve = [&](size_t bytes) { char* p = ws + off; off = (off + bytes + 255) & ~(size_t)255; return p; };
    u16* xbf    = (u16*)carve((size_t)RTOT * DM * 2);        // 4 MB
    u16* WT3    = (u16*)carve((size_t)3 * DM * DM * 2);      // 1.5 MB (WqT,WkT,WvT)
    u16* WoT    = (u16*)carve((size_t)DM * DM * 2);          // 0.5 MB
    u16* Ebf    = (u16*)carve((size_t)LSEQ * PDH * 2);       // 0.25 MB
    u16* qbuf   = (u16*)carve((size_t)NB * NH * LSEQ * PDH * 2); // 4 MB
    u16* kbuf   = (u16*)carve((size_t)NB * NH * LSEQ * PDH * 2); // 4 MB
    u16* vTbuf  = (u16*)carve((size_t)NB * NH * PDH * LSEQ * 2); // 4 MB
    u16* attnbf = (u16*)carve((size_t)RTOT * DM * 2);        // 4 MB

    // 1) precision conversion / weight transpose
    cvt_bf16_kernel<<<(RTOT * DM + 255) / 256, 256, 0, stream>>>(x, xbf, RTOT * DM);
    cvt_bf16_kernel<<<(LSEQ * PDH + 255) / 256, 256, 0, stream>>>(E, Ebf, LSEQ * PDH);
    transpose_cvt_kernel<<<(DM * DM) / 256, 256, 0, stream>>>(Wq, WT3);
    transpose_cvt_kernel<<<(DM * DM) / 256, 256, 0, stream>>>(Wk, WT3 + (size_t)DM * DM);
    transpose_cvt_kernel<<<(DM * DM) / 256, 256, 0, stream>>>(Wv, WT3 + (size_t)2 * DM * DM);
    transpose_cvt_kernel<<<(DM * DM) / 256, 256, 0, stream>>>(Wo, WoT);

    // 2) QKV projections: tiles = (4096/128 row-groups) x (512/16 cols) x 3
    qkv_proj_kernel<<<dim3(DM / 16, RTOT / 128, 3), 256, 0, stream>>>(
        xbf, WT3, bq, bk, bv, qbuf, kbuf, vTbuf);

    // 3) flash attention with relative bias: (128 q-blocks / 8 waves) x (B*H)
    attn_kernel<<<dim3((LSEQ / 16) / 8, NB * NH), 256, 0, stream>>>(
        qbuf, kbuf, vTbuf, Ebf, attnbf);

    // 4) output projection -> fp32 d_out
    out_proj_kernel<<<dim3(DM / 16, RTOT / 128), 256, 0, stream>>>(attnbf, WoT, bo, out);
}
